// TreeCRFLoss_65412351918487
// MI455X (gfx1250) — matrix operand order repaired and checked by reading, not compiled, not run on hardware
//
#include <hip/hip_runtime.h>

// ---------------------------------------------------------------------------
// TreeCRF (CKY inside) for MI455X / gfx1250.
//   B=128, N=41 (n=40 fenceposts), L=128 labels.
// DP over widths w=2..40, all spans of a width in parallel.
// phase1: per-span 128xKx128 exp-domain matmul via v_wmma_f32_16x16x4_f32,
//         inputs staged global->LDS with GLOBAL_LOAD_ASYNC_TO_LDS_B128.
// phase2: batched GEMM (M=spans, K=16384, N=128) via v_wmma_f32_16x16x32_bf16
//         against pre-swizzled exp(T-mt); K-chunked (128) double-buffered LDS
//         staging with async copies + s_wait_asynccnt.
// chart (110MB) + exp(T) (4MB) stay resident in the 192MB L2.
// ---------------------------------------------------------------------------

#define BSZ 128
#define NN  41
#define NF  40      // n
#define LL  128
#define KPAD 40     // max splits (39) padded; multiple of 4
#define CH  1024    // span chunk for P buffer (32MB bf16)
#define KC  128     // phase-2 K chunk staged in LDS

typedef __bf16 bf16;
typedef __attribute__((ext_vector_type(16))) __bf16 v16bf;
typedef __attribute__((ext_vector_type(8)))  float  v8f;
typedef __attribute__((ext_vector_type(2)))  float  v2f;
typedef __attribute__((ext_vector_type(4)))  int    i4;

union FragB16 { v16bf v; uint4 u[2]; };

// ---------------- async global->LDS staging helpers ------------------------
#if __has_builtin(__builtin_amdgcn_global_load_async_to_lds_b128)
#define HAVE_ASYNC 1
typedef __attribute__((address_space(1))) i4 as1_i4;   // global int4
typedef __attribute__((address_space(3))) i4 as3_i4;   // LDS int4
#endif

static __device__ __forceinline__ void cp16(void* l, const void* g) {
#ifdef HAVE_ASYNC
    __builtin_amdgcn_global_load_async_to_lds_b128((as1_i4*)g, (as3_i4*)l, 0, 0);
#else
    *(uint4*)l = *(const uint4*)g;
#endif
}

static __device__ __forceinline__ void wait_async() {
#ifdef HAVE_ASYNC
# if __has_builtin(__builtin_amdgcn_s_wait_asynccnt)
    __builtin_amdgcn_s_wait_asynccnt(0);
# else
    asm volatile("s_wait_asynccnt 0" ::: "memory");
# endif
#endif
}

// ------------------------------ reductions ---------------------------------
__global__ void tcrf_max1(const float* __restrict__ tr, float* __restrict__ part, int ntot) {
    __shared__ float sh[256];
    int t = threadIdx.x;
    float mx = -3.0e38f;
    for (int i = blockIdx.x * 256 + t; i < ntot; i += 256 * 256) mx = fmaxf(mx, tr[i]);
    sh[t] = mx; __syncthreads();
    for (int o = 128; o > 0; o >>= 1) { if (t < o) sh[t] = fmaxf(sh[t], sh[t + o]); __syncthreads(); }
    if (t == 0) part[blockIdx.x] = sh[0];
}

__global__ void tcrf_max2(const float* __restrict__ part, float* __restrict__ mtp) {
    __shared__ float sh[256];
    int t = threadIdx.x;
    sh[t] = part[t]; __syncthreads();
    for (int o = 128; o > 0; o >>= 1) { if (t < o) sh[t] = fmaxf(sh[t], sh[t + o]); __syncthreads(); }
    if (t == 0) mtp[0] = sh[0];
}

// exp(T - mt), pre-swizzled into WMMA B-fragment order:
// element (k, A) -> et[ ((k>>5)*128 + A)*32 + (k&31) ]   (k = x*128+y)
// => a 128-deep K-chunk is one CONTIGUOUS 32KB block (async-copy friendly).
__global__ void tcrf_et(const float* __restrict__ tr, const float* __restrict__ mtp,
                        bf16* __restrict__ et) {
    int e = blockIdx.x * 256 + threadIdx.x;      // < 128^3
    float mt = mtp[0];
    int k = e >> 7, A = e & 127;
    size_t dst = ((size_t)(k >> 5) * 128 + A) * 32 + (size_t)(k & 31);
    et[dst] = (bf16)__expf(tr[e] - mt);
}

// width-1 diagonal: chart[b,d,d+1,:] = scores + start_transitions
__global__ void tcrf_init(float* __restrict__ chart, const float* __restrict__ scores,
                          const float* __restrict__ st) {
    int blk = blockIdx.x;                 // b*NF + d
    int b = blk / NF, d = blk - b * NF;
    int t = threadIdx.x;
    size_t idx = ((((size_t)b) * NN + d) * NN + (d + 1)) * LL + t;
    chart[idx] = scores[idx] + st[t];
}

// ------------------------------- phase 1 -----------------------------------
// One workgroup (256 thr / 8 waves) per span.
__global__ __launch_bounds__(256) void tcrf_phase1(
    const float* __restrict__ chart, const float* __restrict__ mtp,
    bf16* __restrict__ P, float* __restrict__ offb, int w, int S, int base)
{
    extern __shared__ __align__(16) char smemraw[];
    float* ELS = (float*)smemraw;        // [KPAD][128]  left  (k-major)
    float* ERS = ELS + KPAD * 128;       // [KPAD][128]  right (k-major)
    float* PR  = ERS + KPAD * 128;       // [16384] pair -> ltmp
    float* mlS = PR + 16384;             // [128]
    float* mrS = mlS + 128;              // [128]
    float* red = mrS + 128;              // [256]

    const int t = threadIdx.x;
    const int sp = base + blockIdx.x;
    const int b = sp / S, s = sp - (sp / S) * S;
    const int i = s, j = s + w, K = w - 1;

    // ---- async-stage left/right chart blocks into LDS (16B granules) ----
    const float* gl = chart + ((((size_t)b) * NN + i) * NN + (i + 1)) * LL;   // contiguous K*512B
    const float* gr = chart + ((((size_t)b) * NN + (i + 1)) * NN + j) * LL;   // K rows, stride NN*LL
    const int nchunk = K * 32;            // 16B units per matrix
    for (int idx = t; idx < nchunk; idx += 256) {
        int kk = idx >> 5, c = (idx & 31) << 2;   // c: float offset within row
        cp16(ELS + kk * 128 + c, gl + (size_t)kk * LL + c);
        cp16(ERS + kk * 128 + c, gr + (size_t)kk * NN * LL + c);
    }
    wait_async();
    __syncthreads();

    // ---- per-column max + exp normalize (EL) / (ER), zero-pad rows ----
    if (t < 128) {
        int x = t;
        float mx = -3.0e38f;
        for (int kk = 0; kk < K; ++kk) mx = fmaxf(mx, ELS[kk * 128 + x]);
        mlS[x] = mx;
        for (int kk = 0; kk < K; ++kk) ELS[kk * 128 + x] = __expf(ELS[kk * 128 + x] - mx);
        for (int kk = K; kk < KPAD; ++kk) ELS[kk * 128 + x] = 0.f;
    } else {
        int y = t - 128;
        float mx = -3.0e38f;
        for (int kk = 0; kk < K; ++kk) mx = fmaxf(mx, ERS[kk * 128 + y]);
        mrS[y] = mx;
        for (int kk = 0; kk < K; ++kk) ERS[kk * 128 + y] = __expf(ERS[kk * 128 + y] - mx);
        for (int kk = K; kk < KPAD; ++kk) ERS[kk * 128 + y] = 0.f;
    }
    __syncthreads();

    const int ln = t & 31, wv = t >> 5;
#if __has_builtin(__builtin_amdgcn_wmma_f32_16x16x4_f32)
    {
        const int lane15 = ln & 15, hi = ln >> 4;
        const int xr = wv * 16 + lane15;             // A row (x)
        for (int ty = 0; ty < 8; ++ty) {
            v8f acc = {};
            const int yc = ty * 16 + lane15;         // B column (y)
            for (int kb = 0; kb < KPAD; kb += 4) {
                int k0 = kb + (hi << 1);
                v2f a, bb;
                a.x  = ELS[k0 * 128 + xr];
                a.y  = ELS[(k0 + 1) * 128 + xr];
                bb.x = ERS[k0 * 128 + yc];
                bb.y = ERS[(k0 + 1) * 128 + yc];
                acc = __builtin_amdgcn_wmma_f32_16x16x4_f32(
                    false, a, false, bb, (short)0, acc, false, false);
            }
            for (int r = 0; r < 8; ++r)
                PR[(wv * 16 + (hi << 3) + r) * 128 + yc] = acc[r];
        }
    }
#else
    for (int idx = t; idx < 16384; idx += 256) {
        int x = idx >> 7, y = idx & 127;
        float acc = 0.f;
        for (int kk = 0; kk < KPAD; ++kk) acc += ELS[kk * 128 + x] * ERS[kk * 128 + y];
        PR[idx] = acc;
    }
#endif
    __syncthreads();

    // ltmp = log(pair) + ml + mr ; mm = max
    float lmax = -3.0e38f;
    for (int idx = t; idx < 16384; idx += 256) {
        float lt = __logf(PR[idx]) + mlS[idx >> 7] + mrS[idx & 127];
        PR[idx] = lt;
        lmax = fmaxf(lmax, lt);
    }
    red[t] = lmax; __syncthreads();
    for (int o = 128; o > 0; o >>= 1) { if (t < o) red[t] = fmaxf(red[t], red[t + o]); __syncthreads(); }
    float mm = red[0];

    bf16* dst = P + (size_t)blockIdx.x * 16384;
    for (int idx = t; idx < 16384; idx += 256)
        dst[idx] = (bf16)__expf(PR[idx] - mm);
    if (t == 0) offb[blockIdx.x] = mm + mtp[0];
}

// ------------------------------- phase 2 -----------------------------------
// C[m,A] = sum_k P[m,k]*et[k,A], K=16384 in KC=128 chunks, double-buffered
// async LDS staging. Workgroup = 32-row M-tile x full N=128; 8 waves, each
// owns one 16-col N-tile and both 16-row M-blocks.
__global__ __launch_bounds__(256) void tcrf_phase2(
    float* __restrict__ chart, const float* __restrict__ scores,
    const bf16* __restrict__ P, const bf16* __restrict__ et,
    const float* __restrict__ offb, int w, int S, int base, int cnt)
{
    extern __shared__ __align__(16) char smem2[];
    bf16* etS[2]; bf16* pS[2];
    etS[0] = (bf16*)smem2;                 // [4][128][32] = KC x 128 cols
    etS[1] = etS[0] + KC * 128;
    pS[0]  = etS[1] + KC * 128;            // [32][KC]
    pS[1]  = pS[0] + 32 * KC;

    const int t = threadIdx.x;
    const int ln = t & 31;
    const int wv = t >> 5;                 // N-tile 0..7
    const int lane15 = ln & 15;
    const int hi = ln >> 4;
    const int row0 = blockIdx.x * 32;      // chunk-local base row

    const bool v0 = (row0 + lane15) < cnt;
    const bool v1 = (row0 + 16 + lane15) < cnt;
    const int n0 = wv * 16 + lane15;
    const uint4 zz = make_uint4(0u, 0u, 0u, 0u);

    // stage chunk kc into (etB, pB): et chunk is contiguous 32KB; P chunk is
    // 32 rows x 256B.
    auto stage = [&](bf16* etB, bf16* pB, int kc) {
        const bf16* etG = et + (size_t)(kc >> 5) * (128 * 32);
        for (int idx = t; idx < (KC * 128) / 8; idx += 256)          // 2048
            cp16(etB + idx * 8, etG + idx * 8);
        for (int idx = t; idx < (32 * KC) / 8; idx += 256) {         // 512
            int row = idx >> 4, c = (idx & 15) << 3;
            int gr2 = row0 + row; if (gr2 >= cnt) gr2 = cnt - 1;
            cp16(pB + row * KC + c, P + (size_t)gr2 * 16384 + kc + c);
        }
    };

    v8f acc0 = {}, acc1 = {};
    stage(etS[0], pS[0], 0);
    const int NIT = 16384 / KC;            // 128
    for (int it = 0; it < NIT; ++it) {
        const int cur = it & 1;
        wait_async();
        __syncthreads();
        if (it + 1 < NIT) stage(etS[cur ^ 1], pS[cur ^ 1], (it + 1) * KC);

        const bf16* etB = etS[cur];
        const bf16* pB  = pS[cur];
        for (int kbl = 0; kbl < 4; ++kbl) {
            FragB16 bb, a0, a1;
            const uint4* qb = (const uint4*)(etB + ((kbl * 128 + n0) * 32 + (hi << 3)));
            bb.u[0] = qb[0]; bb.u[1] = qb[2];
            const uint4* qa0 = (const uint4*)(pB + (lane15 * KC + kbl * 32 + (hi << 3)));
            a0.u[0] = qa0[0]; a0.u[1] = qa0[2];
            const uint4* qa1 = (const uint4*)(pB + ((16 + lane15) * KC + kbl * 32 + (hi << 3)));
            a1.u[0] = qa1[0]; a1.u[1] = qa1[2];
            if (!v0) { a0.u[0] = zz; a0.u[1] = zz; }
            if (!v1) { a1.u[0] = zz; a1.u[1] = zz; }
            acc0 = __builtin_amdgcn_wmma_f32_16x16x32_bf16(
                false, a0.v, false, bb.v, (short)0, acc0, false, false);
            acc1 = __builtin_amdgcn_wmma_f32_16x16x32_bf16(
                false, a1.v, false, bb.v, (short)0, acc1, false, false);
        }
    }

    // epilogue: chart[b,i,i+w,A] = log(acc) + (mm + mt) + scores
    for (int mb = 0; mb < 2; ++mb) {
        v8f acc = mb ? acc1 : acc0;
        for (int r = 0; r < 8; ++r) {
            int m = row0 + mb * 16 + (hi << 3) + r;
            if (m < cnt) {
                int sp = base + m;
                int b = sp / S, s = sp - (sp / S) * S;
                int col = wv * 16 + lane15;
                float val = __logf(acc[r]) + offb[m];
                size_t idx = ((((size_t)b) * NN + s) * NN + (s + w)) * LL + col;
                chart[idx] = val + scores[idx];
            }
        }
    }
}

// ------------------------------- final -------------------------------------
__global__ void tcrf_final(const float* __restrict__ chart, float* __restrict__ out) {
    __shared__ float sh[128];
    int b = blockIdx.x, t = threadIdx.x;
    float v = chart[((((size_t)b) * NN + 0) * NN + NF) * LL + t];
    sh[t] = v; __syncthreads();
    for (int o = 64; o > 0; o >>= 1) { if (t < o) sh[t] = fmaxf(sh[t], sh[t + o]); __syncthreads(); }
    float mx = sh[0]; __syncthreads();
    sh[t] = __expf(v - mx); __syncthreads();
    for (int o = 64; o > 0; o >>= 1) { if (t < o) sh[t] += sh[t + o]; __syncthreads(); }
    if (t == 0) out[b] = __logf(sh[0]) + mx;
}

// ------------------------------- launch ------------------------------------
extern "C" void kernel_launch(void* const* d_in, const int* in_sizes, int n_in,
                              void* d_out, int out_size, void* d_ws, size_t ws_size,
                              hipStream_t stream) {
    const float* scores = (const float*)d_in[0];
    // d_in[1] = mask (unused by the reference computation)
    const float* trans  = (const float*)d_in[2];
    const float* startt = (const float*)d_in[3];

    char* wsb = (char*)d_ws;
    size_t off = 0;
    float* chart = (float*)(wsb + off); off += (size_t)BSZ * NN * NN * LL * 4;   // 110 MB
    bf16*  etswz = (bf16*)(wsb + off);  off += (size_t)LL * LL * LL * 2;          // 4 MB
    float* mtp   = (float*)(wsb + off); off += 256;
    float* part  = (float*)(wsb + off); off += 256 * 4;
    float* offb  = (float*)(wsb + off); off += (size_t)CH * 4;
    bf16*  Pbuf  = (bf16*)(wsb + off);  off += (size_t)CH * LL * LL * 2;          // 32 MB

    const int NTR = LL * LL * LL;
    tcrf_max1<<<256, 256, 0, stream>>>(trans, part, NTR);
    tcrf_max2<<<1, 256, 0, stream>>>(part, mtp);
    tcrf_et<<<NTR / 256, 256, 0, stream>>>(trans, mtp, etswz);
    tcrf_init<<<BSZ * NF, LL, 0, stream>>>(chart, scores, startt);

    const size_t smem1 = (size_t)(KPAD * 128 * 2 + 16384 + 128 + 128 + 256) * 4;   // ~106 KB
    const size_t smem2 = (size_t)(2 * KC * 128 + 2 * 32 * KC) * 2;                 // 80 KB

    for (int w = 2; w <= NF; ++w) {
        int S = NF - w + 1;                // spans of this width per batch
        int total = BSZ * S;
        for (int base = 0; base < total; base += CH) {
            int cnt = (total - base) < CH ? (total - base) : CH;
            tcrf_phase1<<<cnt, 256, smem1, stream>>>(chart, mtp, Pbuf, offb, w, S, base);
            tcrf_phase2<<<(cnt + 31) / 32, 256, smem2, stream>>>(chart, scores, Pbuf, etswz,
                                                                 offb, w, S, base, cnt);
        }
    }
    tcrf_final<<<BSZ, LL, 0, stream>>>(chart, (float*)d_out);
}
